// SelfAttention_19035295056621
// MI455X (gfx1250) — matrix operand order repaired and checked by reading, not compiled, not run on hardware
//
#include <hip/hip_runtime.h>
#include <hip/hip_bf16.h>
#include <math.h>

typedef _Float16 h16;
typedef __attribute__((ext_vector_type(16))) _Float16 v16h;
typedef __attribute__((ext_vector_type(8)))  _Float16 v8h;
typedef __attribute__((ext_vector_type(8)))  float    v8f;

#define B_ 8
#define S_ 2048
#define E_ 512

// ---------------- WMMA fragment loads (gfx1250 wave32 layouts, ISA 7.12.2) ----
// A (16x32 f16): lane = {m = lane&15, half = lane>>4}; elems 0..7 -> K = half*8 + i,
// elems 8..15 -> K = 16 + half*8 + (i-8).  => two contiguous 16B chunks.
__device__ __forceinline__ v16h load_frag_a(const h16* __restrict__ tile, int ld, int k0) {
  const int lane = threadIdx.x & 31;
  const int m = lane & 15, lh = lane >> 4;
  const h16* p = tile + (long long)m * ld + k0 + lh * 8;
  v8h lo = *(const v8h*)(p);
  v8h hi = *(const v8h*)(p + 16);
  v16h r;
#pragma unroll
  for (int i = 0; i < 8; ++i) { r[i] = lo[i]; r[i + 8] = hi[i]; }
  return r;
}

// B (32x16 f16), operand given N-major (Bt[n][k]): lane = {n = lane&15, half},
// elems 0..7 -> K = half*8 + i, elems 8..15 -> K = 16 + half*8 + (i-8).
__device__ __forceinline__ v16h load_frag_b(const h16* __restrict__ tile, int ld, int k0) {
  const int lane = threadIdx.x & 31;
  const int n = lane & 15, lh = lane >> 4;
  const h16* p = tile + (long long)n * ld + k0 + lh * 8;
  v8h lo = *(const v8h*)(p);
  v8h hi = *(const v8h*)(p + 16);
  v16h r;
#pragma unroll
  for (int i = 0; i < 8; ++i) { r[i] = lo[i]; r[i + 8] = hi[i]; }
  return r;
}

// ---------------- Generic batched WMMA GEMM: C = A[M,K] * Bt[N,K]^T ----------
// MODE 0: f16 C[m*ldc+n]
// MODE 1: f16 transposed C[n*ldc+m]          (used to build V^T)
// MODE 2: f16 C = scale*acc, mask==0 -> -inf (attention scores)
// MODE 3: f32 C = acc + bias[n]              (final output projection)
// Block: 256 threads = 8 waves as 4(m) x 2(n); wave tile 32x64; block tile 128x128.
// All M,N,K used here are exact multiples of 128/128/32 -> no bounds checks.
template <int MODE>
__global__ __launch_bounds__(256)
void gemm_wmma(const h16* __restrict__ A, long long strideA, int lda,
               const h16* __restrict__ Bt, long long strideB, int ldb,
               void* __restrict__ Cv, long long strideC, int ldc,
               int Kdim, float scale,
               const int* __restrict__ mask, long long strideM,
               const float* __restrict__ bias)
{
  const int z = blockIdx.z;
  const h16* Ab = A + (long long)z * strideA;
  const h16* Bb = Bt + (long long)z * strideB;

  const int wave = threadIdx.x >> 5;
  const int wm = wave & 3, wn = wave >> 2;
  const int m0 = blockIdx.y * 128 + wm * 32;
  const int n0 = blockIdx.x * 128 + wn * 64;

  const v8f vzero = {0.f, 0.f, 0.f, 0.f, 0.f, 0.f, 0.f, 0.f};
  v8f acc[2][4];
#pragma unroll
  for (int t = 0; t < 2; ++t)
#pragma unroll
    for (int u = 0; u < 4; ++u) acc[t][u] = vzero;

  const h16* At0 = Ab + (long long)m0 * lda;
  const h16* At1 = Ab + (long long)(m0 + 16) * lda;

  for (int k0 = 0; k0 < Kdim; k0 += 32) {
    if (k0 + 32 < Kdim) {            // pull next K-slab toward L2 (global_prefetch_b8)
      __builtin_prefetch(At0 + k0 + 32, 0, 1);
      __builtin_prefetch(Bb + (long long)n0 * ldb + k0 + 32, 0, 1);
    }
    v16h a0 = load_frag_a(At0, lda, k0);
    v16h a1 = load_frag_a(At1, lda, k0);
    v16h b[4];
#pragma unroll
    for (int u = 0; u < 4; ++u)
      b[u] = load_frag_b(Bb + (long long)(n0 + u * 16) * ldb, ldb, k0);
#pragma unroll
    for (int u = 0; u < 4; ++u) {
      acc[0][u] = __builtin_amdgcn_wmma_f32_16x16x32_f16(
          false, a0, false, b[u], (short)0, acc[0][u], false, false);
      acc[1][u] = __builtin_amdgcn_wmma_f32_16x16x32_f16(
          false, a1, false, b[u], (short)0, acc[1][u], false, false);
    }
  }

  // C/D layout: VGPR r, lanes 0-15: (M=r, N=lane); lanes 16-31: (M=8+r, N=lane-16)
  const int lane = threadIdx.x & 31;
  const int lh = lane >> 4, nn = lane & 15;
#pragma unroll
  for (int t = 0; t < 2; ++t) {
#pragma unroll
    for (int u = 0; u < 4; ++u) {
#pragma unroll
      for (int r = 0; r < 8; ++r) {
        const int m = m0 + t * 16 + lh * 8 + r;
        const int n = n0 + u * 16 + nn;
        float val = acc[t][u][r];
        if constexpr (MODE == 2) {
          val *= scale;
          if (mask[(long long)z * strideM + (long long)m * ldc + n] == 0) val = -1e30f;
          ((h16*)Cv)[(long long)z * strideC + (long long)m * ldc + n] = (h16)val;
        } else if constexpr (MODE == 1) {
          ((h16*)Cv)[(long long)z * strideC + (long long)n * ldc + m] = (h16)val;
        } else if constexpr (MODE == 3) {
          ((float*)Cv)[(long long)z * strideC + (long long)m * ldc + n] = val + bias[n];
        } else {
          ((h16*)Cv)[(long long)z * strideC + (long long)m * ldc + n] = (h16)val;
        }
      }
    }
  }
}

// ---------------- RoPE + f32->f16 cast for q, k, v ---------------------------
__global__ __launch_bounds__(256)
void rope_cast(const float* __restrict__ q, const float* __restrict__ k,
               const float* __restrict__ v,
               h16* __restrict__ qr, h16* __restrict__ kr, h16* __restrict__ vh)
{
  const long long idx = (long long)blockIdx.x * 256 + threadIdx.x; // one (row, pair)
  const int e2 = E_ / 2;
  const long long rowp = idx / e2;     // b*S + s
  const int t = (int)(idx % e2);
  const int s = (int)(rowp % S_);
  // theta_t = 10000^(-2*(2t)/E)
  const float ex = (-2.0f * (float)(2 * t)) / (float)E_;
  const float theta = __expf(ex * 9.2103403719761836f); // ln(10000)
  float sn, cs;
  sincosf((float)s * theta, &sn, &cs);
  const long long base = rowp * E_ + 2 * t;
  float xo = q[base], xe = q[base + 1];
  qr[base]     = (h16)(xo * cs - xe * sn);
  qr[base + 1] = (h16)(xo * sn + xe * cs);
  xo = k[base]; xe = k[base + 1];
  kr[base]     = (h16)(xo * cs - xe * sn);
  kr[base + 1] = (h16)(xo * sn + xe * cs);
  vh[base]     = (h16)v[base];
  vh[base + 1] = (h16)v[base + 1];
}

__global__ __launch_bounds__(256)
void cast_weights(const float* __restrict__ Wq, const float* __restrict__ Wk,
                  const float* __restrict__ Wv, const float* __restrict__ Wo,
                  h16* __restrict__ wq, h16* __restrict__ wk,
                  h16* __restrict__ wv, h16* __restrict__ wo, int n)
{
  const int i = blockIdx.x * 256 + threadIdx.x;
  if (i < n) {
    wq[i] = (h16)Wq[i]; wk[i] = (h16)Wk[i];
    wv[i] = (h16)Wv[i]; wo[i] = (h16)Wo[i];
  }
}

// ---------------- Row softmax over S_ f16 scores, in place -------------------
__global__ __launch_bounds__(256)
void softmax_rows(h16* __restrict__ scores)
{
  const long long row = blockIdx.x;
  h16* p = scores + row * (long long)S_;
  const int tid = threadIdx.x;
  const int lane = tid & 31, wave = tid >> 5;
  __shared__ float smax[8];
  __shared__ float ssum[8];

  float vals[8];
  float m = -3.0e38f;
#pragma unroll
  for (int i = 0; i < 8; ++i) { vals[i] = (float)p[tid + i * 256]; m = fmaxf(m, vals[i]); }
#pragma unroll
  for (int off = 16; off >= 1; off >>= 1) m = fmaxf(m, __shfl_xor(m, off, 32));
  if (lane == 0) smax[wave] = m;
  __syncthreads();
  float rmax = smax[0];
#pragma unroll
  for (int w = 1; w < 8; ++w) rmax = fmaxf(rmax, smax[w]);

  float s = 0.f;
#pragma unroll
  for (int i = 0; i < 8; ++i) { vals[i] = __expf(vals[i] - rmax); s += vals[i]; }
#pragma unroll
  for (int off = 16; off >= 1; off >>= 1) s += __shfl_xor(s, off, 32);
  if (lane == 0) ssum[wave] = s;
  __syncthreads();
  float rsum = 0.f;
#pragma unroll
  for (int w = 0; w < 8; ++w) rsum += ssum[w];
  const float inv = 1.0f / rsum;
#pragma unroll
  for (int i = 0; i < 8; ++i) p[tid + i * 256] = (h16)(vals[i] * inv);
}

// ---------------- Host-side orchestration ------------------------------------
extern "C" void kernel_launch(void* const* d_in, const int* in_sizes, int n_in,
                              void* d_out, int out_size, void* d_ws, size_t ws_size,
                              hipStream_t stream)
{
  (void)in_sizes; (void)n_in; (void)out_size; (void)ws_size;
  const float* q    = (const float*)d_in[0];
  const float* k    = (const float*)d_in[1];
  const float* v    = (const float*)d_in[2];
  const int*   mask = (const int*)  d_in[3];
  const float* Wq   = (const float*)d_in[4];
  const float* Wk   = (const float*)d_in[5];
  const float* Wv   = (const float*)d_in[6];
  const float* Wo   = (const float*)d_in[7];
  const float* bo   = (const float*)d_in[8];

  h16* ws = (h16*)d_ws;
  const long long P   = (long long)B_ * S_ * E_;   // 8,388,608 elems
  const long long sSD = (long long)S_ * E_;
  const long long sSS = (long long)S_ * S_;

  h16* qr   = ws;                 // roped q (f16)        [B,S,E]
  h16* kr   = ws + P;             // roped k (f16)        [B,S,E]
  h16* vh   = ws + 2 * P;         // v cast  (f16)        [B,S,E]
  h16* Qp   = ws + 3 * P;         // Q proj               [B,S,D]
  h16* Kp   = ws + 4 * P;         // K proj               [B,S,D]
  h16* VpT  = ws + 5 * P;         // V proj, transposed   [B,D,S]
  h16* scor = ws + 6 * P;         // scores / probs       [B,S,S]
  h16* wq16 = ws + 6 * P + (long long)B_ * sSS;
  h16* wk16 = wq16 + (long long)E_ * E_;
  h16* wv16 = wk16 + (long long)E_ * E_;
  h16* wo16 = wv16 + (long long)E_ * E_;
  h16* attn = qr;                 // reuse: qr is dead after the Q projection

  cast_weights<<<dim3((E_ * E_ + 255) / 256), dim3(256), 0, stream>>>(
      Wq, Wk, Wv, Wo, wq16, wk16, wv16, wo16, E_ * E_);

  rope_cast<<<dim3((unsigned)((long long)B_ * S_ * (E_ / 2) / 256)), dim3(256), 0, stream>>>(
      q, k, v, qr, kr, vh);

  // Q = rope(q) @ Wq^T   (Bt = Wq, N-major)
  gemm_wmma<0><<<dim3(E_ / 128, S_ / 128, B_), dim3(256), 0, stream>>>(
      qr, sSD, E_, wq16, 0, E_, Qp, sSD, E_, E_, 1.f, nullptr, 0, nullptr);
  // K = rope(k) @ Wk^T
  gemm_wmma<0><<<dim3(E_ / 128, S_ / 128, B_), dim3(256), 0, stream>>>(
      kr, sSD, E_, wk16, 0, E_, Kp, sSD, E_, E_, 1.f, nullptr, 0, nullptr);
  // V^T[d][s] = (v @ Wv^T)[s][d]  (transposed store so PV GEMM gets N-major B)
  gemm_wmma<1><<<dim3(E_ / 128, S_ / 128, B_), dim3(256), 0, stream>>>(
      vh, sSD, E_, wv16, 0, E_, VpT, sSD, S_, E_, 1.f, nullptr, 0, nullptr);
  // scores = (Q K^T) / sqrt(D), mask==0 -> -inf   (Bt = K rows, N-major)
  gemm_wmma<2><<<dim3(S_ / 128, S_ / 128, B_), dim3(256), 0, stream>>>(
      Qp, sSD, E_, Kp, sSD, E_, scor, sSS, S_, E_,
      0.044194173824159216f /* 1/sqrt(512) */, mask, sSS, nullptr);

  softmax_rows<<<dim3(B_ * S_), dim3(256), 0, stream>>>(scor);

  // attn = P @ V   (Bt = V^T, N-major over head dim, K over kv)
  gemm_wmma<0><<<dim3(E_ / 128, S_ / 128, B_), dim3(256), 0, stream>>>(
      scor, sSS, S_, VpT, sSD, S_, attn, sSD, E_, S_, 1.f, nullptr, 0, nullptr);
  // out = attn @ Wo^T + bo  (f32 output)
  gemm_wmma<3><<<dim3(E_ / 128, S_ / 128, B_), dim3(256), 0, stream>>>(
      attn, sSD, E_, wo16, 0, E_, d_out, sSD, E_, E_, 1.f, nullptr, 0, bo);
}
